// Detr_Transformer_26293789786914
// MI455X (gfx1250) — matrix operand order repaired and checked by reading, not compiled, not run on hardware
//
#include <hip/hip_runtime.h>
#include <hip/hip_bf16.h>

typedef __attribute__((ext_vector_type(16))) _Float16 v16h;
typedef __attribute__((ext_vector_type(8)))  _Float16 v8h;
typedef __attribute__((ext_vector_type(8)))  float    v8f;

#define R_TOT 16660
#define RPAD  16672   // multiple of 32 for full 32x32 macro-tiles
#define NHEAD 8

// row -> (scale, index-within-scale); rows are scale-major concat of src0..src3
__device__ __forceinline__ void rowinfo(int r, int& l, int& idx) {
  if (r < 12544)      { l = 0; idx = r; }
  else if (r < 15680) { l = 1; idx = r - 12544; }
  else if (r < 16464) { l = 2; idx = r - 15680; }
  else                { l = 3; idx = r - 16464; }
}

// ---------------- WMMA fragment loaders (f16, K-step 32) ----------------
// A (16x32, MxK): lane m%16 holds row; lane-group g holds 16 consecutive K:
//   halves 0..15 -> K = kb + 16*g + 0..15
__device__ __forceinline__ v16h load_a_frag(const _Float16* __restrict__ base,
                                            int row, int ldk, int kb, int g) {
  const _Float16* p = base + (size_t)row * ldk + kb;
  return *reinterpret_cast<const v16h*>(p + 16 * g);
}
// B (32x16, KxN) loaded from pre-transposed Wt[N,K]: lane n%16 holds column;
// lane-group g holds 16 consecutive K starting at kb + 16*g (same K order as A)
__device__ __forceinline__ v16h load_b_frag(const _Float16* __restrict__ base,
                                            int col, int ldk, int kb, int g) {
  return *reinterpret_cast<const v16h*>(base + (size_t)col * ldk + kb + 16 * g);
}

#define WMMA_F16(A_, B_, C_) \
  __builtin_amdgcn_wmma_f32_16x16x32_f16(false, (A_), false, (B_), (short)0, (C_), false, false)

// ---- generic WMMA GEMM: C[M,N] = A_f16[M,K] @ Wt_f16[N,K]^T + bias ----
// One wave computes a 32x32 macro-tile: 4 WMMAs per K-step from 2 A-frags + 2
// B-frags -> 2 b128 loads per WMMA (2x the arithmetic intensity of 16x16/wave).
__global__ __launch_bounds__(256) void gemm_kernel(
    const _Float16* __restrict__ A, const _Float16* __restrict__ Bt,
    const float* __restrict__ bias,
    float* __restrict__ Cf, _Float16* __restrict__ Ch,
    int M, int N, int K, int relu)
{
  int wave = (int)((blockIdx.x * blockDim.x + threadIdx.x) >> 5);
  int lane = (int)(threadIdx.x & 31);
  int tilesN = N >> 5;
  int numTiles = (M >> 5) * tilesN;
  if (wave >= numTiles) return;              // wave-uniform: EXEC stays all-ones
  int tm = (wave / tilesN) << 5;
  int tn = (wave % tilesN) << 5;
  int g  = lane >> 4;
  int m0 = tm + (lane & 15), m1 = m0 + 16;
  int n0 = tn + (lane & 15), n1 = n0 + 16;
  v8f c00 = {}, c01 = {}, c10 = {}, c11 = {};
  for (int kb = 0; kb < K; kb += 32) {
    v16h a0 = load_a_frag(A,  m0, K, kb, g);
    v16h a1 = load_a_frag(A,  m1, K, kb, g);
    v16h b0 = load_b_frag(Bt, n0, K, kb, g);
    v16h b1 = load_b_frag(Bt, n1, K, kb, g);
    c00 = WMMA_F16(a0, b0, c00);
    c01 = WMMA_F16(a0, b1, c01);
    c10 = WMMA_F16(a1, b0, c10);
    c11 = WMMA_F16(a1, b1, c11);
  }
  float bz0 = bias[n0];
  float bz1 = bias[n1];
  #pragma unroll
  for (int r8 = 0; r8 < 8; ++r8) {
    int row0 = tm + r8 + 8 * g;         // C/D layout: row = r + 8g, col = lane&15
    int row1 = row0 + 16;
    float v00 = c00[r8] + bz0, v01 = c01[r8] + bz1;
    float v10 = c10[r8] + bz0, v11 = c11[r8] + bz1;
    if (relu) {
      v00 = fmaxf(v00, 0.f); v01 = fmaxf(v01, 0.f);
      v10 = fmaxf(v10, 0.f); v11 = fmaxf(v11, 0.f);
    }
    size_t o00 = (size_t)row0 * N + n0, o01 = (size_t)row0 * N + n1;
    size_t o10 = (size_t)row1 * N + n0, o11 = (size_t)row1 * N + n1;
    if (Cf) { Cf[o00] = v00; Cf[o01] = v01; Cf[o10] = v10; Cf[o11] = v11; }
    else    { Ch[o00] = (_Float16)v00; Ch[o01] = (_Float16)v01;
              Ch[o10] = (_Float16)v10; Ch[o11] = (_Float16)v11; }
  }
}

// ---------------- weight transpose + f32->f16: W[K,N] -> WT[N,K] ----------------
__global__ __launch_bounds__(256) void wtrans_kernel(const float* __restrict__ W,
                                                     _Float16* __restrict__ WT,
                                                     int K, int N)
{
  size_t i = (size_t)blockIdx.x * blockDim.x + threadIdx.x;
  if (i >= (size_t)K * N) return;
  int k = (int)(i / (size_t)N), n = (int)(i % (size_t)N);
  WT[(size_t)n * K + k] = (_Float16)W[i];
}

// ---------------- build f16 activation matrix from the 4 src tensors ----------------
__global__ __launch_bounds__(256) void buildx_kernel(
    const float* __restrict__ s0, const float* __restrict__ s1,
    const float* __restrict__ s2, const float* __restrict__ s3,
    _Float16* __restrict__ X16)
{
  size_t i = (size_t)blockIdx.x * blockDim.x + threadIdx.x;
  if (i >= (size_t)RPAD * 512) return;
  int r = (int)(i >> 9), cch = (int)(i & 511);
  float v = 0.f;
  if (r < R_TOT) {
    int l, idx; rowinfo(r, l, idx);
    const float* sp = (l == 0) ? s0 : (l == 1) ? s1 : (l == 2) ? s2 : s3;
    v = sp[(size_t)idx * 512 + cch];
  }
  X16[i] = (_Float16)v;
}

// ---------------- softmax over the 16 (S*K) logits per (row, head), in place ----------------
__global__ __launch_bounds__(256) void softmax_kernel(float* __restrict__ ATT)
{
  int t = blockIdx.x * blockDim.x + threadIdx.x;
  if (t >= R_TOT * NHEAD) return;
  float* p = ATT + (size_t)(t >> 3) * 128 + (size_t)(t & 7) * 16;
  float m = p[0];
  #pragma unroll
  for (int i = 1; i < 16; ++i) m = fmaxf(m, p[i]);
  float e[16], s = 0.f;
  #pragma unroll
  for (int i = 0; i < 16; ++i) { e[i] = expf(p[i] - m); s += e[i]; }
  float inv = 1.f / s;
  #pragma unroll
  for (int i = 0; i < 16; ++i) p[i] = e[i] * inv;
}

// ---------------- multi-scale deformable bilinear sampling ----------------
// one wave per (row, head); lane owns 2 channels of DH=64; scalar math is wave-uniform
__global__ __launch_bounds__(256) void msdeform_kernel(
    const float* __restrict__ V, const float* __restrict__ OFF,
    const float* __restrict__ ATT,
    const float* __restrict__ ref0, const float* __restrict__ ref1,
    const float* __restrict__ ref2, const float* __restrict__ ref3,
    _Float16* __restrict__ ACCh)
{
  int wid  = (int)((blockIdx.x * blockDim.x + threadIdx.x) >> 5);
  int lane = (int)(threadIdx.x & 31);
  if (wid >= RPAD * NHEAD) return;
  int r = wid >> 3, h = wid & 7;
  int dh = lane * 2;
  float a0 = 0.f, a1 = 0.f;
  if (r < R_TOT) {
    const int HWr[4] = {3136, 784, 196, 49};
    const int WLs[4] = {56, 28, 14, 7};
    const int VB [4] = {0, 12544, 15680, 16464};
    int l, idx; rowinfo(r, l, idx);
    const float* refp = (l == 0) ? ref0 : (l == 1) ? ref1 : (l == 2) ? ref2 : ref3;
    float rx = refp[(size_t)idx * 2 + 0];
    float ry = refp[(size_t)idx * 2 + 1];
    int bi = idx / HWr[l];
    #pragma unroll
    for (int s = 0; s < 4; ++s) {
      const int wl = WLs[s], hl = WLs[s];
      const int vb = VB[s] + bi * hl * wl;
      #pragma unroll
      for (int k = 0; k < 4; ++k) {
        float at = ATT[(size_t)r * 128 + h * 16 + s * 4 + k];
        float ox = OFF[(size_t)r * 256 + (((h * 4 + s) * 4 + k) * 2) + 0];
        float oy = OFF[(size_t)r * 256 + (((h * 4 + s) * 4 + k) * 2) + 1];
        float x = (rx + ox / (float)wl) * (float)wl - 0.5f;
        float y = (ry + oy / (float)hl) * (float)hl - 0.5f;
        float x0f = floorf(x), y0f = floorf(y);
        int   x0 = (int)x0f,  y0 = (int)y0f;
        float wx1 = x - x0f, wx0 = 1.f - wx1;
        float wy1 = y - y0f, wy0 = 1.f - wy1;
        int   cx[4] = {x0, x0 + 1, x0,     x0 + 1};
        int   cy[4] = {y0, y0,     y0 + 1, y0 + 1};
        float cw[4] = {wx0 * wy0, wx1 * wy0, wx0 * wy1, wx1 * wy1};
        #pragma unroll
        for (int c = 0; c < 4; ++c) {
          int xi = cx[c], yi = cy[c];
          if (xi >= 0 && xi < wl && yi >= 0 && yi < hl) {
            const float* vp = V + (size_t)(vb + yi * wl + xi) * 512 + h * 64 + dh;
            float w = at * cw[c];
            a0 += w * vp[0];
            a1 += w * vp[1];
          }
        }
      }
    }
  }
  ACCh[(size_t)r * 512 + h * 64 + dh]     = (_Float16)a0;
  ACCh[(size_t)r * 512 + h * 64 + dh + 1] = (_Float16)a1;
}

// ---------------- layernorm 1: X1 = LN(q + O); one wave per row ----------------
__global__ __launch_bounds__(256) void ln1_kernel(
    const float* __restrict__ s0, const float* __restrict__ s1,
    const float* __restrict__ s2, const float* __restrict__ s3,
    const float* __restrict__ O,
    const float* __restrict__ g, const float* __restrict__ be,
    float* __restrict__ X1, _Float16* __restrict__ X1h)
{
  int wid  = (int)((blockIdx.x * blockDim.x + threadIdx.x) >> 5);
  int lane = (int)(threadIdx.x & 31);
  if (wid >= R_TOT) return;
  int r = wid, l, idx; rowinfo(r, l, idx);
  const float* q = ((l == 0) ? s0 : (l == 1) ? s1 : (l == 2) ? s2 : s3) + (size_t)idx * 512;
  const float* o = O + (size_t)r * 512;
  int c0 = lane * 16;
  float x[16], s = 0.f, ss = 0.f;
  #pragma unroll
  for (int j = 0; j < 16; ++j) { float v = q[c0 + j] + o[c0 + j]; x[j] = v; s += v; ss += v * v; }
  for (int m = 16; m > 0; m >>= 1) { s += __shfl_xor(s, m); ss += __shfl_xor(ss, m); }
  float mu  = s * (1.f / 512.f);
  float var = fmaxf(ss * (1.f / 512.f) - mu * mu, 0.f);
  float rstd = rsqrtf(var + 1e-5f);
  #pragma unroll
  for (int j = 0; j < 16; ++j) {
    float y = (x[j] - mu) * rstd * g[c0 + j] + be[c0 + j];
    X1 [(size_t)r * 512 + c0 + j] = y;
    X1h[(size_t)r * 512 + c0 + j] = (_Float16)y;
  }
}

// ---------------- layernorm 2: out = LN(X1 + F2), stored at permuted (batch-major) row ----------------
__global__ __launch_bounds__(256) void ln2_kernel(
    const float* __restrict__ X1, const float* __restrict__ F2,
    const float* __restrict__ g, const float* __restrict__ be,
    float* __restrict__ out)
{
  int wid  = (int)((blockIdx.x * blockDim.x + threadIdx.x) >> 5);
  int lane = (int)(threadIdx.x & 31);
  if (wid >= R_TOT) return;
  int r = wid, l, idx; rowinfo(r, l, idx);
  const int HWr[4] = {3136, 784, 196, 49};
  const int QB [4] = {0, 3136, 3920, 4116};
  int bi = idx / HWr[l], pix = idx % HWr[l];
  size_t orow = (size_t)bi * 4165 + QB[l] + pix;
  int c0 = lane * 16;
  float x[16], s = 0.f, ss = 0.f;
  #pragma unroll
  for (int j = 0; j < 16; ++j) {
    float v = X1[(size_t)r * 512 + c0 + j] + F2[(size_t)r * 512 + c0 + j];
    x[j] = v; s += v; ss += v * v;
  }
  for (int m = 16; m > 0; m >>= 1) { s += __shfl_xor(s, m); ss += __shfl_xor(ss, m); }
  float mu  = s * (1.f / 512.f);
  float var = fmaxf(ss * (1.f / 512.f) - mu * mu, 0.f);
  float rstd = rsqrtf(var + 1e-5f);
  #pragma unroll
  for (int j = 0; j < 16; ++j) {
    float y = (x[j] - mu) * rstd * g[c0 + j] + be[c0 + j];
    out[orow * 512 + c0 + j] = y;
  }
}

extern "C" void kernel_launch(void* const* d_in, const int* in_sizes, int n_in,
                              void* d_out, int out_size, void* d_ws, size_t ws_size,
                              hipStream_t stream) {
  (void)in_sizes; (void)n_in; (void)out_size; (void)ws_size;
  const float* s0 = (const float*)d_in[0];  const float* r0 = (const float*)d_in[1];
  const float* s1 = (const float*)d_in[2];  const float* r1 = (const float*)d_in[3];
  const float* s2 = (const float*)d_in[4];  const float* r2 = (const float*)d_in[5];
  const float* s3 = (const float*)d_in[6];  const float* r3 = (const float*)d_in[7];
  const float* Wv    = (const float*)d_in[8];   const float* bv    = (const float*)d_in[9];
  const float* Woff  = (const float*)d_in[10];  const float* boff  = (const float*)d_in[11];
  const float* Wattn = (const float*)d_in[12];  const float* battn = (const float*)d_in[13];
  const float* Wo    = (const float*)d_in[14];  const float* bo    = (const float*)d_in[15];
  const float* W1    = (const float*)d_in[16];  const float* b1    = (const float*)d_in[17];
  const float* W2    = (const float*)d_in[18];  const float* b2    = (const float*)d_in[19];
  const float* g1    = (const float*)d_in[20];  const float* be1   = (const float*)d_in[21];
  const float* g2    = (const float*)d_in[22];  const float* be2   = (const float*)d_in[23];
  float* out = (float*)d_out;

  // ---- workspace layout (deterministic, lifetime-aliased) ----
  char* ws = (char*)d_ws;
  size_t off = 0;
  auto alloc = [&](size_t bytes) { size_t p = off; off = (off + bytes + 255) & ~(size_t)255; return p; };
  size_t o_WvT   = alloc((size_t)512 * 512 * 2);
  size_t o_WoffT = alloc((size_t)256 * 512 * 2);
  size_t o_WatT  = alloc((size_t)128 * 512 * 2);
  size_t o_WoT   = alloc((size_t)512 * 512 * 2);
  size_t o_W1T   = alloc((size_t)2048 * 512 * 2);
  size_t o_W2T   = alloc((size_t)512 * 2048 * 2);
  size_t arena   = off;                               // early-phase arena
  size_t o_X16 = alloc((size_t)RPAD * 512 * 2);
  size_t o_V   = alloc((size_t)RPAD * 512 * 4);
  size_t o_OFF = alloc((size_t)RPAD * 256 * 4);
  size_t o_ATT = alloc((size_t)RPAD * 128 * 4);
  size_t o_FFh = arena;                               // FFN acts reuse arena (68.3MB <= 76.8MB)
  size_t o_ACC = alloc((size_t)RPAD * 512 * 2);
  size_t o_O   = alloc((size_t)RPAD * 512 * 4);       // also reused for F2
  size_t o_X1  = alloc((size_t)RPAD * 512 * 4);
  size_t o_X1h = alloc((size_t)RPAD * 512 * 2);

  _Float16* WvT   = (_Float16*)(ws + o_WvT);
  _Float16* WoffT = (_Float16*)(ws + o_WoffT);
  _Float16* WatT  = (_Float16*)(ws + o_WatT);
  _Float16* WoT   = (_Float16*)(ws + o_WoT);
  _Float16* W1T   = (_Float16*)(ws + o_W1T);
  _Float16* W2T   = (_Float16*)(ws + o_W2T);
  _Float16* X16   = (_Float16*)(ws + o_X16);
  float*    V     = (float*)   (ws + o_V);
  float*    OFF   = (float*)   (ws + o_OFF);
  float*    ATT   = (float*)   (ws + o_ATT);
  _Float16* FFh   = (_Float16*)(ws + o_FFh);
  _Float16* ACCh  = (_Float16*)(ws + o_ACC);
  float*    O     = (float*)   (ws + o_O);
  float*    F2    = O;
  float*    X1    = (float*)   (ws + o_X1);
  _Float16* X1h   = (_Float16*)(ws + o_X1h);

  auto cdiv = [](long a, long b) { return (int)((a + b - 1) / b); };
  auto gblocks = [&](int N) { return cdiv((long)(RPAD / 32) * (N / 32), 8); };

  // 1) weight transpose + f16 convert
  wtrans_kernel<<<cdiv(512L * 512, 256), 256, 0, stream>>>(Wv,    WvT,   512, 512);
  wtrans_kernel<<<cdiv(512L * 256, 256), 256, 0, stream>>>(Woff,  WoffT, 512, 256);
  wtrans_kernel<<<cdiv(512L * 128, 256), 256, 0, stream>>>(Wattn, WatT,  512, 128);
  wtrans_kernel<<<cdiv(512L * 512, 256), 256, 0, stream>>>(Wo,    WoT,   512, 512);
  wtrans_kernel<<<cdiv(512L * 2048, 256), 256, 0, stream>>>(W1,   W1T,   512, 2048);
  wtrans_kernel<<<cdiv(2048L * 512, 256), 256, 0, stream>>>(W2,   W2T,   2048, 512);

  // 2) f16 activation matrix (zero-padded rows)
  buildx_kernel<<<cdiv((long)RPAD * 512, 256), 256, 0, stream>>>(s0, s1, s2, s3, X16);

  // 3) shared projections (WMMA GEMMs, 32x32 macro-tiles)
  gemm_kernel<<<gblocks(512), 256, 0, stream>>>(X16, WvT,   bv,    V,   nullptr, RPAD, 512, 512, 0);
  gemm_kernel<<<gblocks(256), 256, 0, stream>>>(X16, WoffT, boff,  OFF, nullptr, RPAD, 256, 512, 0);
  gemm_kernel<<<gblocks(128), 256, 0, stream>>>(X16, WatT,  battn, ATT, nullptr, RPAD, 128, 512, 0);

  // 4) softmax over S*K per (row, head)
  softmax_kernel<<<cdiv((long)R_TOT * NHEAD, 256), 256, 0, stream>>>(ATT);

  // 5) deformable sampling -> ACC (f16, pad rows zeroed)
  msdeform_kernel<<<RPAD, 256, 0, stream>>>(V, OFF, ATT, r0, r1, r2, r3, ACCh);

  // 6) output projection, residual + LN1
  gemm_kernel<<<gblocks(512), 256, 0, stream>>>(ACCh, WoT, bo, O, nullptr, RPAD, 512, 512, 0);
  ln1_kernel<<<cdiv(R_TOT, 8), 256, 0, stream>>>(s0, s1, s2, s3, O, g1, be1, X1, X1h);

  // 7) FFN (WMMA, ReLU fused, f16 intermediate), residual + LN2 -> d_out (permuted rows)
  gemm_kernel<<<gblocks(2048), 256, 0, stream>>>(X1h, W1T, b1, nullptr, FFh, RPAD, 2048, 512, 1);
  gemm_kernel<<<gblocks(512),  256, 0, stream>>>(FFh, W2T, b2, F2, nullptr, RPAD, 512, 2048, 0);
  ln2_kernel<<<cdiv(R_TOT, 8), 256, 0, stream>>>(X1, F2, g2, be2, out);
}
